// DynamicGraphIpaFrameDenoiser_49959059587660
// MI455X (gfx1250) — compile-verified, matrix operand order checked by reading, not run on hardware
//
#include <hip/hip_runtime.h>

typedef __attribute__((ext_vector_type(16))) __bf16 v16bf;
typedef __attribute__((ext_vector_type(8)))  float  v8f;
typedef unsigned short u16;

// ---------------- helpers ----------------
__device__ __forceinline__ u16 f2bf(float f) {
  unsigned int u = __float_as_uint(f);
  unsigned int lsb = (u >> 16) & 1u;
  return (u16)((u + 0x7fffu + lsb) >> 16);   // RNE f32 -> bf16
}

__device__ __forceinline__ void atomicMaxF(float* addr, float val) {
  unsigned int* ua = (unsigned int*)addr;
  unsigned int old = *ua;
  while (__uint_as_float(old) < val) {
    unsigned int assumed = old;
    old = atomicCAS(ua, assumed, __float_as_uint(val));
    if (old == assumed) break;
  }
}

// ---------------- small utility kernels ----------------
__global__ void k_fill(float* p, float v, int n) {
  int i = blockIdx.x * blockDim.x + threadIdx.x;
  if (i < n) p[i] = v;
}

// f32 [M,Cin] -> bf16 [M,Cpad] (zero-padded)
__global__ void k_cvt(const float* __restrict__ in, u16* __restrict__ out,
                      int M, int Cin, int Cpad) {
  int i = blockIdx.x * blockDim.x + threadIdx.x;
  if (i >= M * Cpad) return;
  int r = i / Cpad, c = i - r * Cpad;
  out[i] = (c < Cin) ? f2bf(in[(size_t)r * Cin + c]) : (u16)0;
}

// transpose+convert weight rows [r0, r0+Krows) of W[*, ldN] -> Wt[Ncols, Kpad] bf16
__global__ void k_wt(const float* __restrict__ W, int ldN, int r0, int Krows,
                     int Ncols, int Kpad, u16* __restrict__ out) {
  int i = blockIdx.x * blockDim.x + threadIdx.x;
  if (i >= Ncols * Kpad) return;
  int n = i / Kpad, kk = i - n * Kpad;
  out[i] = (kk < Krows) ? f2bf(W[(size_t)(r0 + kk) * ldN + n]) : (u16)0;
}

// concat(latent[128], nf[256]) -> bf16 [N,384]
__global__ void k_latnf(const float* __restrict__ lat, const float* __restrict__ nf,
                        u16* __restrict__ out, int N) {
  int i = blockIdx.x * blockDim.x + threadIdx.x;
  if (i >= N * 384) return;
  int r = i / 384, c = i - r * 384;
  float v = (c < 128) ? lat[(size_t)r * 128 + c] : nf[(size_t)r * 256 + (c - 128)];
  out[i] = f2bf(v);
}

__global__ void k_quat2rot(const float* __restrict__ q4, float* __restrict__ R, int N) {
  int n = blockIdx.x * blockDim.x + threadIdx.x;
  if (n >= N) return;
  float w = q4[n*4+0], x = q4[n*4+1], y = q4[n*4+2], z = q4[n*4+3];
  float inv = rsqrtf(w*w + x*x + y*y + z*z + 1e-30f);
  w *= inv; x *= inv; y *= inv; z *= inv;
  float* r = R + (size_t)n * 9;
  r[0] = 1.f - 2.f*(y*y + z*z); r[1] = 2.f*(x*y - w*z); r[2] = 2.f*(x*z + w*y);
  r[3] = 2.f*(x*y + w*z); r[4] = 1.f - 2.f*(x*x + z*z); r[5] = 2.f*(y*z - w*x);
  r[6] = 2.f*(x*z - w*y); r[7] = 2.f*(y*z + w*x); r[8] = 1.f - 2.f*(x*x + y*y);
}

// local points [N, HP*3] -> global: g = R@p + t
__global__ void k_pts(const float* __restrict__ loc, const float* __restrict__ R,
                      const float* __restrict__ t, float* __restrict__ g, int N, int HP) {
  int i = blockIdx.x * blockDim.x + threadIdx.x;
  if (i >= N * HP) return;
  int n = i / HP, j = i - n * HP;
  const float* p  = loc + (size_t)n * HP * 3 + j * 3;
  const float* Rn = R + (size_t)n * 9;
  const float* tn = t + (size_t)n * 3;
  float x = p[0], y = p[1], z = p[2];
  float* o = g + (size_t)n * HP * 3 + j * 3;
  o[0] = Rn[0]*x + Rn[1]*y + Rn[2]*z + tn[0];
  o[1] = Rn[3]*x + Rn[4]*y + Rn[5]*z + tn[1];
  o[2] = Rn[6]*x + Rn[7]*y + Rn[8]*z + tn[2];
}

// ---------------- WMMA GEMM: C[M,Nc] = A[M,K](bf16) x Bt[Nc,K](bf16)^T + epilogue ----
// flags bit0 = relu, bit1 = edge mode (+= addS[src[m]] + addD[dst[m]])
__global__ void k_gemm(const u16* __restrict__ A, const u16* __restrict__ Bt,
                       const float* __restrict__ bias, float* __restrict__ C,
                       int M, int Nc, int K, int flags,
                       const float* __restrict__ addS, const float* __restrict__ addD,
                       const int* __restrict__ srcIdx, const int* __restrict__ dstIdx) {
  const int lane = threadIdx.x & 31;
  const int wave = blockIdx.x * (blockDim.x >> 5) + (threadIdx.x >> 5);
  const int tilesN = Nc >> 4;
  const int tileM = wave / tilesN;
  const int tileN = wave - tileM * tilesN;
  if (tileM >= (M >> 4)) return;                 // wave-uniform: EXEC stays full
  const int m0 = tileM << 4, n0 = tileN << 4;
  const int l16 = lane & 15;
  const int hi  = (lane >> 4) << 3;              // K sub-offset per lane half
  const u16* arow = A  + (size_t)(m0 + l16) * K;
  const u16* brow = Bt + (size_t)(n0 + l16) * K;
  v8f acc = {0.f, 0.f, 0.f, 0.f, 0.f, 0.f, 0.f, 0.f};
  for (int k0 = 0; k0 < K; k0 += 32) {
    union { v16bf v; uint4 q[2]; } fa, fb;
    fa.q[0] = *(const uint4*)(arow + k0 + hi);
    fa.q[1] = *(const uint4*)(arow + k0 + hi + 16);
    fb.q[0] = *(const uint4*)(brow + k0 + hi);
    fb.q[1] = *(const uint4*)(brow + k0 + hi + 16);
    acc = __builtin_amdgcn_wmma_f32_16x16x32_bf16(false, fa.v, false, fb.v,
                                                  (short)0, acc, false, false);
  }
  const int col = n0 + l16;
  const float bn = bias ? bias[col] : 0.f;
  const int rbase = m0 + ((lane >> 4) << 3);     // D layout: vgpr r -> row r (+8 for hi lanes)
  for (int r = 0; r < 8; ++r) {
    int row = rbase + r;
    float cv = acc[r] + bn;
    if (flags & 2) {
      int s = srcIdx[row], d = dstIdx[row];
      cv += addS[(size_t)s * Nc + col] + addD[(size_t)d * Nc + col];
    }
    if (flags & 1) cv = fmaxf(cv, 0.f);
    C[(size_t)row * Nc + col] = cv;
  }
}

// ---------------- LayerNorm: out = LN(x + res*resScale)*g + b, then *postScale ----
__global__ void k_ln(const float* __restrict__ x, const float* __restrict__ res,
                     const float* __restrict__ resScale,
                     const float* __restrict__ g, const float* __restrict__ b,
                     const float* __restrict__ post, float* __restrict__ out, int C) {
  int row = blockIdx.x, tid = threadIdx.x;
  __shared__ float sb[128];
  const float* xr = x + (size_t)row * C;
  const float* rr = res ? res + (size_t)row * C : nullptr;
  float rs = resScale ? resScale[row] : 1.f;
  float sum = 0.f;
  for (int c = tid; c < C; c += 128) sum += xr[c] + (rr ? rr[c] * rs : 0.f);
  sb[tid] = sum; __syncthreads();
  for (int s = 64; s > 0; s >>= 1) { if (tid < s) sb[tid] += sb[tid + s]; __syncthreads(); }
  float mean = sb[0] / C; __syncthreads();
  float vs = 0.f;
  for (int c = tid; c < C; c += 128) {
    float v = xr[c] + (rr ? rr[c] * rs : 0.f) - mean; vs += v * v;
  }
  sb[tid] = vs; __syncthreads();
  for (int s = 64; s > 0; s >>= 1) { if (tid < s) sb[tid] += sb[tid + s]; __syncthreads(); }
  float rstd = rsqrtf(sb[0] / C + 1e-5f);
  float ps = post ? post[row] : 1.f;
  for (int c = tid; c < C; c += 128) {
    float v = xr[c] + (rr ? rr[c] * rs : 0.f);
    out[(size_t)row * C + c] = ((v - mean) * rstd * g[c] + b[c]) * ps;
  }
}

// ---------------- IPA attention kernels ----------------
// one thread per (edge, head): logits + atomic segment max
__global__ void k_logits(const float* __restrict__ q, const float* __restrict__ k,
                         const float* __restrict__ qpg, const float* __restrict__ kpg,
                         const float* __restrict__ z, const float* __restrict__ Wb,
                         const float* __restrict__ bb, const float* __restrict__ gam,
                         const float* __restrict__ mask,
                         const int* __restrict__ src, const int* __restrict__ dst,
                         float* __restrict__ a, float* __restrict__ amax, int nE) {
  int i = blockIdx.x * blockDim.x + threadIdx.x;
  if (i >= nE * 8) return;
  int e = i >> 3, h = i & 7;
  int s = src[e], d = dst[e];
  const float* qr = q + (size_t)d * 256 + h * 32;
  const float* kr = k + (size_t)s * 256 + h * 32;
  float acc = 0.f;
  for (int c = 0; c < 32; ++c) acc += qr[c] * kr[c];
  acc *= 0.10206207261596575f;                   // sqrt(1/(3*32))
  const float* zr = z + (size_t)e * 128;
  float bias = bb[h];
  for (int c = 0; c < 128; ++c) bias += zr[c] * Wb[c * 8 + h];
  acc += 0.5773502691896258f * bias;             // sqrt(1/3)
  float gv = gam[h];
  float sp = (gv > 20.f) ? gv : log1pf(expf(gv));
  float hw = sp * 0.09622504486493764f;          // sqrt(1/(3*36))
  const float* qp = qpg + (size_t)d * 192 + h * 24;
  const float* kp = kpg + (size_t)s * 192 + h * 24;
  float dd = 0.f;
  for (int j = 0; j < 24; ++j) { float t = qp[j] - kp[j]; dd += t * t; }
  acc -= 0.5f * hw * dd;
  acc += (mask[s] - 1.f) * 1e5f;
  a[i] = acc;
  atomicMaxF(&amax[d * 8 + h], acc);
}

__global__ void k_expsum(float* __restrict__ a, const float* __restrict__ amax,
                         float* __restrict__ den, const int* __restrict__ dst, int nE) {
  int i = blockIdx.x * blockDim.x + threadIdx.x;
  if (i >= nE * 8) return;
  int e = i >> 3, h = i & 7;
  int d = dst[e];
  float ae = expf(a[i] - amax[d * 8 + h]);
  a[i] = ae;
  atomicAdd(&den[d * 8 + h], ae);
}

// one block per edge: weighted scatter of v / vpg / z into dst segments
__global__ void k_scatter(const float* __restrict__ a, const float* __restrict__ den,
                          const int* __restrict__ src, const int* __restrict__ dst,
                          const float* __restrict__ v, const float* __restrict__ vpg,
                          const float* __restrict__ z,
                          float* __restrict__ o, float* __restrict__ opt,
                          float* __restrict__ opair) {
  int e = blockIdx.x, tid = threadIdx.x;
  int s = src[e], d = dst[e];
  __shared__ float w[8];
  if (tid < 8) w[tid] = a[e * 8 + tid] / (den[d * 8 + tid] + 1e-9f);
  __syncthreads();
  for (int j = tid; j < 256; j += 256)
    atomicAdd(&o[(size_t)d * 256 + j], w[j >> 5] * v[(size_t)s * 256 + j]);
  for (int j = tid; j < 288; j += 256)
    atomicAdd(&opt[(size_t)d * 288 + j], w[j / 36] * vpg[(size_t)s * 288 + j]);
  for (int j = tid; j < 1024; j += 256)
    atomicAdd(&opair[(size_t)d * 1024 + j], w[j >> 7] * z[(size_t)e * 128 + (j & 127)]);
}

// one block per node: build feat = [o(256) | optl(288) | optn(96) | opair(1024)]
__global__ void k_feat(const float* __restrict__ o, const float* __restrict__ opt,
                       const float* __restrict__ opair, const float* __restrict__ R,
                       const float* __restrict__ t, float* __restrict__ feat) {
  int n = blockIdx.x, tid = threadIdx.x;
  float* fr = feat + (size_t)n * 1664;
  const float* Rn = R + (size_t)n * 9;
  float t0 = t[n*3+0], t1 = t[n*3+1], t2 = t[n*3+2];
  for (int j = tid; j < 256; j += 256) fr[j] = o[(size_t)n * 256 + j];
  for (int j = tid; j < 96; j += 256) {           // j = h*12 + p
    float p0 = opt[(size_t)n*288 + j*3+0] - t0;
    float p1 = opt[(size_t)n*288 + j*3+1] - t1;
    float p2 = opt[(size_t)n*288 + j*3+2] - t2;
    float l0 = Rn[0]*p0 + Rn[3]*p1 + Rn[6]*p2;    // R^T @ p
    float l1 = Rn[1]*p0 + Rn[4]*p1 + Rn[7]*p2;
    float l2 = Rn[2]*p0 + Rn[5]*p1 + Rn[8]*p2;
    fr[256 + j*3+0] = l0; fr[256 + j*3+1] = l1; fr[256 + j*3+2] = l2;
    fr[544 + j] = sqrtf(l0*l0 + l1*l1 + l2*l2 + 1e-8f);
  }
  for (int j = tid; j < 1024; j += 256) fr[640 + j] = opair[(size_t)n * 1024 + j];
}

// ---------------- host orchestration ----------------
extern "C" void kernel_launch(void* const* d_in, const int* in_sizes, int n_in,
                              void* d_out, int out_size, void* d_ws, size_t ws_size,
                              hipStream_t stream) {
  (void)in_sizes; (void)out_size; (void)ws_size;
  const int N = 2048, E = 65536, ES = 8192;
  const float* P[80];
  for (int i = 0; i < 80 && i < n_in; ++i) P[i] = (const float*)d_in[i];
  const float* nf    = P[0];
  const float* lat   = P[1];
  const float* efin  = P[2];
  const float* sefin = P[3];
  const float* nsein = P[4];
  const float* quats = P[5];
  const float* trans = P[6];
  const float* mask  = P[7];
  const int* eidx  = (const int*)d_in[8];
  const int* seidx = (const int*)d_in[9];
  const int *esrc = eidx, *edst = eidx + E;
  const int *qsrc = seidx, *qdst = seidx + ES;

  float* out_s   = (float*)d_out;
  float* out_ef  = out_s + (size_t)N * 256;
  float* out_sef = out_ef + (size_t)E * 128;

  char* wsb = (char*)d_ws; size_t off = 0;
  auto alloc = [&](size_t bytes) -> void* {
    void* p = wsb + off; off = (off + bytes + 255) & ~(size_t)255; return p;
  };
  // scratch
  float* Rm      = (float*)alloc((size_t)N * 9 * 4);
  u16*   nf_bf   = (u16*)alloc((size_t)N * 256 * 2);
  u16*   latnf_bf= (u16*)alloc((size_t)N * 384 * 2);
  u16*   ein_bf  = (u16*)alloc((size_t)E * 224 * 2);     // reused for seq phase
  u16*   bufE_bf = (u16*)alloc((size_t)E * 128 * 2);
  float* bufE_f  = (float*)alloc((size_t)E * 128 * 4);
  float* nfS     = (float*)alloc((size_t)N * 128 * 4);
  float* nfD     = (float*)alloc((size_t)N * 128 * 4);
  float* sA      = (float*)alloc((size_t)N * 256 * 4);
  float* sB      = (float*)alloc((size_t)N * 256 * 4);
  float* tmpN    = (float*)alloc((size_t)N * 256 * 4);
  u16*   tmpN_bf = (u16*)alloc((size_t)N * 256 * 2);
  u16*   s_bf    = (u16*)alloc((size_t)N * 256 * 2);
  float* qb      = (float*)alloc((size_t)N * 256 * 4);
  float* kb      = (float*)alloc((size_t)N * 256 * 4);
  float* vb      = (float*)alloc((size_t)N * 256 * 4);
  float* qpl     = (float*)alloc((size_t)N * 192 * 4);
  float* kpl     = (float*)alloc((size_t)N * 192 * 4);
  float* vpl     = (float*)alloc((size_t)N * 288 * 4);
  float* qpg     = (float*)alloc((size_t)N * 192 * 4);
  float* kpg     = (float*)alloc((size_t)N * 192 * 4);
  float* vpg     = (float*)alloc((size_t)N * 288 * 4);
  float* ob      = (float*)alloc((size_t)N * 256 * 4);
  float* optb    = (float*)alloc((size_t)N * 288 * 4);
  float* opairb  = (float*)alloc((size_t)N * 1024 * 4);
  float* featb   = (float*)alloc((size_t)N * 1664 * 4);
  u16*   feat_bf = (u16*)alloc((size_t)N * 1664 * 2);
  float* upd     = (float*)alloc((size_t)N * 256 * 4);
  float* abuf    = (float*)alloc((size_t)E * 8 * 4);
  float* amax    = (float*)alloc((size_t)N * 8 * 4);
  float* den     = (float*)alloc((size_t)N * 8 * 4);
  // transposed bf16 weights
  u16 *eW0a = (u16*)alloc(128*224*2), *eW0s = (u16*)alloc(128*256*2), *eW0d = (u16*)alloc(128*256*2);
  u16 *eW1 = (u16*)alloc(128*128*2), *eW2 = (u16*)alloc(128*128*2);
  u16 *sW0a = (u16*)alloc(128*224*2), *sW0s = (u16*)alloc(128*256*2), *sW0d = (u16*)alloc(128*256*2);
  u16 *sW1 = (u16*)alloc(128*128*2), *sW2 = (u16*)alloc(128*128*2);
  u16 *l2W0 = (u16*)alloc(256*384*2), *l2W1 = (u16*)alloc(256*256*2), *l2W2 = (u16*)alloc(256*256*2);
  u16 *spQ = (u16*)alloc(256*256*2), *spK = (u16*)alloc(256*256*2), *spV = (u16*)alloc(256*256*2);
  u16 *spQP = (u16*)alloc(192*256*2), *spKP = (u16*)alloc(192*256*2), *spVP = (u16*)alloc(288*256*2);
  u16 *spO = (u16*)alloc((size_t)256*1664*2);
  u16 *sqQ = (u16*)alloc(256*256*2), *sqK = (u16*)alloc(256*256*2), *sqV = (u16*)alloc(256*256*2);
  u16 *sqQP = (u16*)alloc(192*256*2), *sqKP = (u16*)alloc(192*256*2), *sqVP = (u16*)alloc(288*256*2);
  u16 *sqO = (u16*)alloc((size_t)256*1664*2);
  u16 *trW0 = (u16*)alloc(256*256*2), *trW1 = (u16*)alloc(256*256*2), *trW2 = (u16*)alloc(256*256*2);

  auto cvt = [&](const float* in, u16* o, int M, int Cin, int Cpad) {
    k_cvt<<<(M * Cpad + 255) / 256, 256, 0, stream>>>(in, o, M, Cin, Cpad);
  };
  auto wt = [&](const float* W, int ldN, int r0, int Krows, int Ncols, int Kpad, u16* o) {
    k_wt<<<(Ncols * Kpad + 255) / 256, 256, 0, stream>>>(W, ldN, r0, Krows, Ncols, Kpad, o);
  };
  auto gemm = [&](const u16* A, const u16* Bt, const float* bias, float* C,
                  int M, int Nc, int K, int flags,
                  const float* aS = nullptr, const float* aD = nullptr,
                  const int* si = nullptr, const int* di = nullptr) {
    int waves = (M >> 4) * (Nc >> 4);
    k_gemm<<<(waves + 3) >> 2, 128, 0, stream>>>(A, Bt, bias, C, M, Nc, K, flags, aS, aD, si, di);
  };
  auto ln = [&](const float* x, const float* res, const float* rsc,
                const float* g, const float* b, const float* post, float* o, int C, int rows) {
    k_ln<<<rows, 128, 0, stream>>>(x, res, rsc, g, b, post, o, C);
  };

  // ---- phase 0: rotations + input/weight conversion ----
  k_quat2rot<<<(N + 255) / 256, 256, 0, stream>>>(quats, Rm, N);
  cvt(nf, nf_bf, N, 256, 256);
  k_latnf<<<(N * 384 + 255) / 256, 256, 0, stream>>>(lat, nf, latnf_bf, N);
  // params (JAX tree order, sorted dict keys): see index map in comments below
  wt(P[12], 128, 0,   200, 128, 224, eW0a);  // edge_mlp W0 split
  wt(P[12], 128, 200, 256, 128, 256, eW0s);
  wt(P[12], 128, 456, 256, 128, 256, eW0d);
  wt(P[14], 128, 0, 128, 128, 128, eW1);
  wt(P[16], 128, 0, 128, 128, 128, eW2);
  wt(P[66], 128, 0,   200, 128, 224, sW0a);  // seq_mlp W0 split
  wt(P[66], 128, 200, 256, 128, 256, sW0s);
  wt(P[66], 128, 456, 256, 128, 256, sW0d);
  wt(P[68], 128, 0, 128, 128, 128, sW1);
  wt(P[70], 128, 0, 128, 128, 128, sW2);
  wt(P[52], 256, 0, 384, 256, 384, l2W0);    // l2n
  wt(P[54], 256, 0, 256, 256, 256, l2W1);
  wt(P[56], 256, 0, 256, 256, 256, l2W2);
  wt(P[44], 256, 0, 256, 256, 256, spQ);     // ipa_sp: q,k,v,qp,kp,vp,o
  wt(P[38], 256, 0, 256, 256, 256, spK);
  wt(P[48], 256, 0, 256, 256, 256, spV);
  wt(P[46], 192, 0, 256, 192, 256, spQP);
  wt(P[40], 192, 0, 256, 192, 256, spKP);
  wt(P[50], 288, 0, 256, 288, 256, spVP);
  wt(P[42], 256, 0, 1664, 256, 1664, spO);
  wt(P[27], 256, 0, 256, 256, 256, sqQ);     // ipa_seq
  wt(P[21], 256, 0, 256, 256, 256, sqK);
  wt(P[31], 256, 0, 256, 256, 256, sqV);
  wt(P[29], 192, 0, 256, 192, 256, sqQP);
  wt(P[23], 192, 0, 256, 192, 256, sqKP);
  wt(P[33], 288, 0, 256, 288, 256, sqVP);
  wt(P[25], 256, 0, 1664, 256, 1664, sqO);
  wt(P[74], 256, 0, 256, 256, 256, trW0);    // trans_mlp
  wt(P[76], 256, 0, 256, 256, 256, trW1);
  wt(P[78], 256, 0, 256, 256, 256, trW2);

  // ---- phase 1: edge MLP (ef) with hoisted node projections ----
  gemm(nf_bf, eW0s, nullptr, nfS, N, 128, 256, 0);
  gemm(nf_bf, eW0d, nullptr, nfD, N, 128, 256, 0);
  cvt(efin, ein_bf, E, 200, 224);
  gemm(ein_bf, eW0a, P[13], bufE_f, E, 128, 224, 3, nfS, nfD, esrc, edst);
  cvt(bufE_f, bufE_bf, E, 128, 128);
  gemm(bufE_bf, eW1, P[15], bufE_f, E, 128, 128, 1);
  cvt(bufE_f, bufE_bf, E, 128, 128);
  gemm(bufE_bf, eW2, P[17], bufE_f, E, 128, 128, 0);
  ln(bufE_f, nullptr, nullptr, P[10], P[11], nullptr, out_ef, 128, E);

  // ---- phase 2: seq edge MLP (sef), residual with seq_edge_features ----
  gemm(nf_bf, sW0s, nullptr, nfS, N, 128, 256, 0);
  gemm(nf_bf, sW0d, nullptr, nfD, N, 128, 256, 0);
  cvt(nsein, ein_bf, ES, 200, 224);
  gemm(ein_bf, sW0a, P[67], bufE_f, ES, 128, 224, 3, nfS, nfD, qsrc, qdst);
  cvt(bufE_f, bufE_bf, ES, 128, 128);
  gemm(bufE_bf, sW1, P[69], bufE_f, ES, 128, 128, 1);
  cvt(bufE_f, bufE_bf, ES, 128, 128);
  gemm(bufE_bf, sW2, P[71], bufE_f, ES, 128, 128, 0);
  ln(bufE_f, sefin, nullptr, P[64], P[65], nullptr, out_sef, 128, ES);

  // ---- phase 3: s0 = ln0(nf + l2n(concat(lat, nf))) ----
  gemm(latnf_bf, l2W0, P[53], tmpN, N, 256, 384, 1);
  cvt(tmpN, tmpN_bf, N, 256, 256);
  gemm(tmpN_bf, l2W1, P[55], tmpN, N, 256, 256, 1);
  cvt(tmpN, tmpN_bf, N, 256, 256);
  gemm(tmpN_bf, l2W2, P[57], tmpN, N, 256, 256, 0);
  ln(nf, tmpN, nullptr, P[58], P[59], nullptr, sA, 256, N);

  // ---- IPA (shared for sp / seq) ----
  auto ipa = [&](const float* sIn, const float* z, const int* si, const int* di, int nE,
                 const u16* Wq, const float* bq, const u16* Wk, const float* bk2,
                 const u16* Wv, const float* bv, const u16* Wqp, const float* bqp,
                 const u16* Wkp, const float* bkp, const u16* Wvp, const float* bvp,
                 const u16* Wo, const float* bo,
                 const float* Wb, const float* bb, const float* gam, float* updOut) {
    cvt(sIn, s_bf, N, 256, 256);
    gemm(s_bf, Wq,  bq,  qb,  N, 256, 256, 0);
    gemm(s_bf, Wk,  bk2, kb,  N, 256, 256, 0);
    gemm(s_bf, Wv,  bv,  vb,  N, 256, 256, 0);
    gemm(s_bf, Wqp, bqp, qpl, N, 192, 256, 0);
    gemm(s_bf, Wkp, bkp, kpl, N, 192, 256, 0);
    gemm(s_bf, Wvp, bvp, vpl, N, 288, 256, 0);
    k_pts<<<(N * 64 + 255) / 256, 256, 0, stream>>>(qpl, Rm, trans, qpg, N, 64);
    k_pts<<<(N * 64 + 255) / 256, 256, 0, stream>>>(kpl, Rm, trans, kpg, N, 64);
    k_pts<<<(N * 96 + 255) / 256, 256, 0, stream>>>(vpl, Rm, trans, vpg, N, 96);
    k_fill<<<(N * 8 + 255) / 256, 256, 0, stream>>>(amax, -3e38f, N * 8);
    hipMemsetAsync(den,    0, (size_t)N * 8 * 4,    stream);
    hipMemsetAsync(ob,     0, (size_t)N * 256 * 4,  stream);
    hipMemsetAsync(optb,   0, (size_t)N * 288 * 4,  stream);
    hipMemsetAsync(opairb, 0, (size_t)N * 1024 * 4, stream);
    k_logits<<<(nE * 8 + 255) / 256, 256, 0, stream>>>(qb, kb, qpg, kpg, z, Wb, bb, gam,
                                                       mask, si, di, abuf, amax, nE);
    k_expsum<<<(nE * 8 + 255) / 256, 256, 0, stream>>>(abuf, amax, den, di, nE);
    k_scatter<<<nE, 256, 0, stream>>>(abuf, den, si, di, vb, vpg, z, ob, optb, opairb);
    k_feat<<<N, 256, 0, stream>>>(ob, optb, opairb, Rm, trans, featb);
    cvt(featb, feat_bf, N, 1664, 1664);
    gemm(feat_bf, Wo, bo, updOut, N, 256, 1664, 0);
  };

  // ---- phase 4: spatial IPA + ln1 ----
  ipa(sA, out_ef, esrc, edst, E,
      spQ, P[45], spK, P[39], spV, P[49], spQP, P[47], spKP, P[41], spVP, P[51],
      spO, P[43], P[35], P[36], P[37], upd);
  ln(sA, upd, mask, P[60], P[61], nullptr, sB, 256, N);

  // ---- phase 5: sequence IPA + ln2 ----
  ipa(sB, out_sef, qsrc, qdst, ES,
      sqQ, P[28], sqK, P[22], sqV, P[32], sqQP, P[30], sqKP, P[24], sqVP, P[34],
      sqO, P[26], P[18], P[19], P[20], upd);
  ln(sB, upd, mask, P[62], P[63], nullptr, sA, 256, N);

  // ---- phase 6: transition MLP + trans_ln, final mask ----
  cvt(sA, s_bf, N, 256, 256);
  gemm(s_bf, trW0, P[75], tmpN, N, 256, 256, 1);
  cvt(tmpN, tmpN_bf, N, 256, 256);
  gemm(tmpN_bf, trW1, P[77], tmpN, N, 256, 256, 1);
  cvt(tmpN, tmpN_bf, N, 256, 256);
  gemm(tmpN_bf, trW2, P[79], tmpN, N, 256, 256, 0);
  ln(sA, tmpN, nullptr, P[72], P[73], mask, out_s, 256, N);
}
